// DilatedTransformerModel_28192165331605
// MI455X (gfx1250) — matrix operand order repaired and checked by reading, not compiled, not run on hardware
//
#include <hip/hip_runtime.h>
#include <hip/hip_bf16.h>

// ---------------------------------------------------------------------------
// DilatedTransformerModel on MI455X (gfx1250, wave32, WMMA).
// Dense GEMMs: v_wmma_f32_16x16x32_f16, 16x64 strip per wave, register
// double-buffered K loop (next-step fragments load while current WMMAs run).
// ---------------------------------------------------------------------------

typedef __attribute__((ext_vector_type(16))) _Float16 v16h;
typedef __attribute__((ext_vector_type(8)))  _Float16 v8h;
typedef __attribute__((ext_vector_type(4)))  _Float16 h4;
typedef __attribute__((ext_vector_type(8)))  float    v8f;

#define NLAYERS 9
#define NHEAD   2
#define ALEN    5
#define DMODEL  128
#define DHID    512
#define HD      64
#define BB      4
#define TT      4096
#define MROWS   (BB*TT)          // 16384
#define INV_SQRT_HD 0.125f

// ------------------------- weight transpose to f16 -------------------------
// WT[n*K + k] = (f16) W[k*N + n]   (B-fragment wants K-major columns)
__global__ __launch_bounds__(256) void k_xpose(const float* __restrict__ W,
                                               _Float16* __restrict__ WT,
                                               int K, int N) {
  int o = blockIdx.x * 256 + threadIdx.x;
  if (o >= K * N) return;
  int n = o / K, k = o % K;
  WT[o] = (_Float16)W[(size_t)k * N + n];
}

// ------------------------------- WMMA GEMM ---------------------------------
// C[M,N] = A[M,K] @ W[K,N] + bias, A row-major f16, BT = W^T (N x K) f16.
// One wave computes a 16x64 strip: 1 A-fragment feeds 4 WMMAs per K-step.
// K loop is register double-buffered for load/WMMA overlap within a wave.
// EPI: 0 store f32   1 relu->f16   2 H += val   3 H += val, SS += val
template<int EPI, int N, int K>
__global__ __launch_bounds__(256) void k_gemm(const _Float16* __restrict__ A,
                                              const _Float16* __restrict__ BT,
                                              const float* __restrict__ bias,
                                              float* __restrict__ Cf,
                                              _Float16* __restrict__ Ch,
                                              float* __restrict__ H,
                                              float* __restrict__ SS) {
  const int lane = threadIdx.x & 31;
  const int wave = threadIdx.x >> 5;
  const int tile = blockIdx.x * 8 + wave;
  const int NB   = N >> 6;              // 64-wide n strips
  const int nb   = tile % NB;
  const int mt   = tile / NB;
  const int r    = lane & 15;

  const int klo = (lane < 16) ? 0 : 16;   // A: one 16-half run per lane (two v8h loads)
  const int bko = (lane < 16) ? 0 : 16;   // B: one 16-half run per lane
  const _Float16* ap = A  + (size_t)(mt * 16 + r) * K + klo;
  const _Float16* bp = BT + (size_t)(nb * 64 + r) * K + bko;

  v8f acc[4];
#pragma unroll
  for (int j = 0; j < 4; ++j) acc[j] = (v8f){};

  // preload K-step 0
  v8h  a0 = *(const v8h*)(ap);
  v8h  a1 = *(const v8h*)(ap + 8);
  v16h b0 = *(const v16h*)(bp);
  v16h b1 = *(const v16h*)(bp + (size_t)16 * K);
  v16h b2 = *(const v16h*)(bp + (size_t)32 * K);
  v16h b3 = *(const v16h*)(bp + (size_t)48 * K);

#pragma unroll
  for (int k0 = 32; k0 <= K; k0 += 32) {
    // prefetch next K-step (compile-time guard: no branch after unroll)
    v8h  na0 = {}, na1 = {};
    v16h nb0 = {}, nb1 = {}, nb2 = {}, nb3 = {};
    if (k0 < K) {
      na0 = *(const v8h*)(ap + k0);
      na1 = *(const v8h*)(ap + k0 + 8);
      nb0 = *(const v16h*)(bp + k0);
      nb1 = *(const v16h*)(bp + (size_t)16 * K + k0);
      nb2 = *(const v16h*)(bp + (size_t)32 * K + k0);
      nb3 = *(const v16h*)(bp + (size_t)48 * K + k0);
    }
    v16h av;
#pragma unroll
    for (int e = 0; e < 8; ++e) { av[e] = a0[e]; av[e + 8] = a1[e]; }
    acc[0] = __builtin_amdgcn_wmma_f32_16x16x32_f16(false, av, false, b0,
                                                    (short)0, acc[0], false, false);
    acc[1] = __builtin_amdgcn_wmma_f32_16x16x32_f16(false, av, false, b1,
                                                    (short)0, acc[1], false, false);
    acc[2] = __builtin_amdgcn_wmma_f32_16x16x32_f16(false, av, false, b2,
                                                    (short)0, acc[2], false, false);
    acc[3] = __builtin_amdgcn_wmma_f32_16x16x32_f16(false, av, false, b3,
                                                    (short)0, acc[3], false, false);
    a0 = na0; a1 = na1; b0 = nb0; b1 = nb1; b2 = nb2; b3 = nb3;
  }

  const int mbase = mt * 16 + ((lane < 16) ? 0 : 8);
#pragma unroll
  for (int j = 0; j < 4; ++j) {
    const int n     = nb * 64 + j * 16 + r;
    const float bvl = bias[n];
#pragma unroll
    for (int g = 0; g < 8; ++g) {
      size_t o = (size_t)(mbase + g) * N + n;
      float val = acc[j][g] + bvl;
      if constexpr (EPI == 0)      Cf[o] = val;
      else if constexpr (EPI == 1) Ch[o] = (_Float16)fmaxf(val, 0.f);
      else if constexpr (EPI == 2) H[o] += val;
      else                         { H[o] += val; SS[o] += val; }
    }
  }
}

// ------------------------------ conv frontend ------------------------------
// conv1 (32,1,5,3) pad_h=2  + pool3(W) + relu.  out c1[b][oc][t][wp], wp<42
__global__ __launch_bounds__(256) void k_conv1(const float* __restrict__ x,
                                               const float* __restrict__ w,
                                               const float* __restrict__ bias,
                                               float* __restrict__ c1) {
  int i  = blockIdx.x * 256 + threadIdx.x;      // 4*32*4096*42 total
  int wp = i % 42;
  int t  = (i / 42) & 4095;
  int oc = (i / (42 * 4096)) & 31;
  int b  = i / (42 * 4096 * 32);
  float wv[15];
#pragma unroll
  for (int j = 0; j < 15; ++j) wv[j] = w[oc * 15 + j];
  float bval = bias[oc];
  float best = -INFINITY;
#pragma unroll
  for (int j = 0; j < 3; ++j) {
    int w0 = wp * 3 + j;
    float a = bval;
#pragma unroll
    for (int kh = 0; kh < 5; ++kh) {
      int tt = t + kh - 2;
      if (tt < 0 || tt >= TT) continue;
      const float* xr = x + ((size_t)b * TT + tt) * 128 + w0;
      a += xr[0] * wv[kh * 3] + xr[1] * wv[kh * 3 + 1] + xr[2] * wv[kh * 3 + 2];
    }
    best = fmaxf(best, a);
  }
  c1[i] = fmaxf(best, 0.f);
}

// conv2 (64,32,1,12) pad 0 + pool3 + relu. out c2[b][oc][t][wp], wp<10
__global__ __launch_bounds__(256) void k_conv2(const float* __restrict__ c1,
                                               const float* __restrict__ w,
                                               const float* __restrict__ bias,
                                               float* __restrict__ c2) {
  __shared__ float sw[384];
  int oc = blockIdx.y;
  for (int i = threadIdx.x; i < 384; i += 256) sw[i] = w[oc * 384 + i];
  __syncthreads();
  int j  = blockIdx.x * 256 + threadIdx.x;      // 4*4096*10 total
  int wp = j % 10;
  int t  = (j / 10) & 4095;
  int b  = j / (10 * 4096);
  float a0, a1, a2;
  a0 = a1 = a2 = bias[oc];
  for (int ic = 0; ic < 32; ++ic) {
    const float* r = c1 + ((size_t)(b * 32 + ic) * TT + t) * 42 + wp * 3;
    float row[14];
#pragma unroll
    for (int q = 0; q < 14; ++q) row[q] = r[q];
    const float* wr = sw + ic * 12;
#pragma unroll
    for (int kw = 0; kw < 12; ++kw) {
      float wk = wr[kw];
      a0 += row[kw] * wk; a1 += row[kw + 1] * wk; a2 += row[kw + 2] * wk;
    }
  }
  float best = fmaxf(fmaxf(a0, a1), a2);
  c2[((size_t)(b * 64 + oc) * TT + t) * 10 + wp] = fmaxf(best, 0.f);
}

// conv3 (128,64,3,6) pad_h=1 + pool3 + relu, writes h[b][t][oc] (transposed)
__global__ __launch_bounds__(256) void k_conv3(const float* __restrict__ c2,
                                               const float* __restrict__ w,
                                               const float* __restrict__ bias,
                                               float* __restrict__ h) {
  __shared__ float sw[1152];
  int oc = blockIdx.y;
  for (int i = threadIdx.x; i < 1152; i += 256) sw[i] = w[oc * 1152 + i];
  __syncthreads();
  int j = blockIdx.x * 256 + threadIdx.x;       // 4*4096 total
  int t = j & 4095;
  int b = j >> 12;
  float a0, a1, a2;
  a0 = a1 = a2 = bias[oc];
  for (int kh = 0; kh < 3; ++kh) {
    int tt = t + kh - 1;
    if (tt < 0 || tt >= TT) continue;
    for (int ic = 0; ic < 64; ++ic) {
      const float* r  = c2 + ((size_t)(b * 64 + ic) * TT + tt) * 10;
      const float* wr = sw + (ic * 3 + kh) * 6;
#pragma unroll
      for (int kw = 0; kw < 6; ++kw) {
        float wk = wr[kw];
        a0 += r[kw] * wk; a1 += r[kw + 1] * wk; a2 += r[kw + 2] * wk;
      }
    }
  }
  float best = fmaxf(fmaxf(a0, a1), a2);
  h[((size_t)b * TT + t) * 128 + oc] = fmaxf(best, 0.f);
}

// ------------------------------- LayerNorm ---------------------------------
// one wave32 per row (D=128, 4 elems/lane), emits f16 (A operand for GEMM)
__global__ __launch_bounds__(256) void k_ln(const float* __restrict__ x,
                                            const float* __restrict__ s,
                                            const float* __restrict__ bb,
                                            _Float16* __restrict__ y) {
  int lane = threadIdx.x & 31;
  int row  = blockIdx.x * 8 + (threadIdx.x >> 5);
  float4 xv = *(const float4*)(x + (size_t)row * 128 + lane * 4);
  float sum = xv.x + xv.y + xv.z + xv.w;
  float sq  = xv.x * xv.x + xv.y * xv.y + xv.z * xv.z + xv.w * xv.w;
#pragma unroll
  for (int off = 16; off > 0; off >>= 1) {
    sum += __shfl_xor(sum, off, 32);
    sq  += __shfl_xor(sq,  off, 32);
  }
  float mean = sum * (1.f / 128.f);
  float var  = sq * (1.f / 128.f) - mean * mean;
  float rstd = rsqrtf(var + 1e-5f);
  int c = lane * 4;
  h4 o;
  o[0] = (_Float16)((xv.x - mean) * rstd * s[c]     + bb[c]);
  o[1] = (_Float16)((xv.y - mean) * rstd * s[c + 1] + bb[c + 1]);
  o[2] = (_Float16)((xv.z - mean) * rstd * s[c + 2] + bb[c + 2]);
  o[3] = (_Float16)((xv.w - mean) * rstd * s[c + 3] + bb[c + 3]);
  *(h4*)(y + (size_t)row * 128 + c) = o;
}

// --------------------------- dilated local attention -----------------------
// one thread per (b,t,head); 5 taps, rel-pos bias, masked softmax; f16 out
__global__ __launch_bounds__(256) void k_attn(const float* __restrict__ q,
                                              const float* __restrict__ k,
                                              const float* __restrict__ v,
                                              const float* __restrict__ Er,
                                              _Float16* __restrict__ out,
                                              int dil) {
  int i = blockIdx.x * 256 + threadIdx.x;       // 4*4096*2 total
  int hh = i & 1;
  int t  = (i >> 1) & 4095;
  int b  = i >> 13;
  size_t base = ((size_t)b * TT + t) * 128 + hh * HD;
  const float* qp = q + base;
  float4 qr[16];
#pragma unroll
  for (int d = 0; d < 16; ++d) qr[d] = *(const float4*)(qp + d * 4);

  float sc[ALEN];
  int   ix[ALEN];
#pragma unroll
  for (int a = 0; a < ALEN; ++a) {
    int off  = t + (a - 2) * dil;
    bool ok  = (off >= 0) && (off < TT);
    int idx  = min(max(off, 0), TT - 1);
    ix[a]    = idx;
    const float* kp = k + ((size_t)b * TT + idx) * 128 + hh * HD;
    const float* ep = Er + (hh * ALEN + a) * HD;
    float d0 = 0.f;
#pragma unroll
    for (int d = 0; d < 16; ++d) {
      float4 kv = *(const float4*)(kp + d * 4);
      float4 ev = *(const float4*)(ep + d * 4);
      d0 += qr[d].x * (kv.x + ev.x) + qr[d].y * (kv.y + ev.y) +
            qr[d].z * (kv.z + ev.z) + qr[d].w * (kv.w + ev.w);
    }
    sc[a] = ok ? d0 * INV_SQRT_HD : -INFINITY;
  }
  float mx = sc[0];
#pragma unroll
  for (int a = 1; a < ALEN; ++a) mx = fmaxf(mx, sc[a]);
  float wa[ALEN], tot = 0.f;
#pragma unroll
  for (int a = 0; a < ALEN; ++a) { wa[a] = __expf(sc[a] - mx); tot += wa[a]; }
  float inv = 1.f / tot;
#pragma unroll
  for (int a = 0; a < ALEN; ++a) wa[a] *= inv;

#pragma unroll
  for (int d = 0; d < 16; ++d) {
    float ox = 0, oy = 0, oz = 0, ow = 0;
#pragma unroll
    for (int a = 0; a < ALEN; ++a) {
      const float* vp = v + ((size_t)b * TT + ix[a]) * 128 + hh * HD + d * 4;
      float4 vv = *(const float4*)vp;
      ox += wa[a] * vv.x; oy += wa[a] * vv.y;
      oz += wa[a] * vv.z; ow += wa[a] * vv.w;
    }
    h4 o; o[0] = (_Float16)ox; o[1] = (_Float16)oy;
          o[2] = (_Float16)oz; o[3] = (_Float16)ow;
    *(h4*)(out + base + d * 4) = o;
  }
}

// ------------------------------- output heads ------------------------------
__global__ __launch_bounds__(256) void k_out(const float* __restrict__ h,
                                             const float* __restrict__ Wout,
                                             const float* __restrict__ bout,
                                             float* __restrict__ out) {
  int i = blockIdx.x * 256 + threadIdx.x;       // 16384
  const float* hr = h + (size_t)i * 128;
  float a0 = bout[0], a1 = bout[1];
#pragma unroll 4
  for (int d = 0; d < 128; ++d) {
    float r = fmaxf(hr[d], 0.f);
    a0 += r * Wout[d * 2];
    a1 += r * Wout[d * 2 + 1];
  }
  out[(size_t)i * 2]     = a0;
  out[(size_t)i * 2 + 1] = a1;
}

__global__ void k_tpool(const float* __restrict__ ss, float* __restrict__ m) {
  int b = blockIdx.x, d = threadIdx.x;          // <<<4,128>>>
  float acc = 0.f;
  for (int t = 0; t < TT; ++t)
    acc += fmaxf(ss[((size_t)b * TT + t) * 128 + d], 0.f);
  m[b * 128 + d] = acc * (1.f / (float)TT);
}

__global__ __launch_bounds__(256) void k_tout(const float* __restrict__ m,
                                              const float* __restrict__ Wt,
                                              const float* __restrict__ bt,
                                              float* __restrict__ tout) {
  int i = blockIdx.x * 256 + threadIdx.x;
  if (i >= BB * 300) return;
  int b = i / 300, j = i % 300;
  float acc = bt[j];
#pragma unroll 4
  for (int d = 0; d < 128; ++d) acc += m[b * 128 + d] * Wt[d * 300 + j];
  tout[i] = acc;
}

// ------------------------------- host driver -------------------------------
extern "C" void kernel_launch(void* const* d_in, const int* in_sizes, int n_in,
                              void* d_out, int out_size, void* d_ws, size_t ws_size,
                              hipStream_t stream) {
  const float* x       = (const float*)d_in[0];
  const float* c1w     = (const float*)d_in[1];
  const float* c1b     = (const float*)d_in[2];
  const float* c2w     = (const float*)d_in[3];
  const float* c2b     = (const float*)d_in[4];
  const float* c3w     = (const float*)d_in[5];
  const float* c3b     = (const float*)d_in[6];
  const float* ln1_s   = (const float*)d_in[7];
  const float* ln1_b   = (const float*)d_in[8];
  const float* ln2_s   = (const float*)d_in[9];
  const float* ln2_b   = (const float*)d_in[10];
  const float* Wq      = (const float*)d_in[11];
  const float* bq      = (const float*)d_in[12];
  const float* Wk      = (const float*)d_in[13];
  const float* bk      = (const float*)d_in[14];
  const float* Wv      = (const float*)d_in[15];
  const float* bv      = (const float*)d_in[16];
  const float* Wo      = (const float*)d_in[17];
  const float* bo      = (const float*)d_in[18];
  const float* Er      = (const float*)d_in[19];
  const float* W1      = (const float*)d_in[20];
  const float* b1      = (const float*)d_in[21];
  const float* W2      = (const float*)d_in[22];
  const float* b2      = (const float*)d_in[23];
  const float* Wout    = (const float*)d_in[24];
  const float* bout    = (const float*)d_in[25];
  const float* Wt      = (const float*)d_in[26];
  const float* bt      = (const float*)d_in[27];
  float* out = (float*)d_out;                        // 32768 + 1200 floats

  char* ws = (char*)d_ws;
  const size_t H_BYTES  = (size_t)MROWS * 128 * 4;   // 8 MB
  const size_t OFF_H    = 0;
  const size_t OFF_SSUM = OFF_H + H_BYTES;
  const size_t OFF_WT   = OFF_SSUM + H_BYTES;
  const size_t LSTRIDE  = 4 * 16384 + 2 * 65536;     // halves per layer = 196608
  const size_t WT_BYTES = (size_t)NLAYERS * LSTRIDE * 2;
  const size_t OFF_POOL = OFF_WT + WT_BYTES;

  float*    h    = (float*)(ws + OFF_H);
  float*    ssum = (float*)(ws + OFF_SSUM);
  _Float16* wT   = (_Float16*)(ws + OFF_WT);
  char*     pool = ws + OFF_POOL;

  // conv-phase overlay
  float* c1buf = (float*)pool;                                   // 88.1 MB
  float* c2buf = (float*)(pool + (size_t)4 * 32 * TT * 42 * 4);  // 41.9 MB
  // transformer-phase overlay (conv bufs dead by then)
  _Float16* y16  = (_Float16*)pool;                         // 4 MB
  float*    qb   = (float*)(pool + 4194304);                // 8 MB
  float*    kb   = (float*)(pool + 4194304 + 8388608);
  float*    vb   = (float*)(pool + 4194304 + 2 * 8388608);
  _Float16* a16  = (_Float16*)(pool + 4194304 + 3 * 8388608);   // 4 MB
  _Float16* a1b  = (_Float16*)(pool + 2 * 4194304 + 3 * 8388608); // 16 MB
  float*    mb   = (float*)(pool + 3 * 4194304 + 3 * 8388608 + 16777216);

  hipMemsetAsync(ssum, 0, H_BYTES, stream);

  // --- weight transposition (f32 -> f16, K-major) for all layers ---
  for (int l = 0; l < NLAYERS; ++l) {
    _Float16* wl = wT + (size_t)l * LSTRIDE;
    k_xpose<<<64, 256, 0, stream>>>(Wq + (size_t)l * 16384, wl,          128, 128);
    k_xpose<<<64, 256, 0, stream>>>(Wk + (size_t)l * 16384, wl + 16384,  128, 128);
    k_xpose<<<64, 256, 0, stream>>>(Wv + (size_t)l * 16384, wl + 32768,  128, 128);
    k_xpose<<<64, 256, 0, stream>>>(Wo + (size_t)l * 16384, wl + 49152,  128, 128);
    k_xpose<<<256, 256, 0, stream>>>(W1 + (size_t)l * 65536, wl + 65536, 128, 512);
    k_xpose<<<256, 256, 0, stream>>>(W2 + (size_t)l * 65536, wl + 131072, 512, 128);
  }

  // --- conv frontend (fused conv + pool3 + relu each) ---
  k_conv1<<<(4 * 32 * TT * 42) / 256, 256, 0, stream>>>(x, c1w, c1b, c1buf);
  k_conv2<<<dim3((4 * TT * 10) / 256, 64), 256, 0, stream>>>(c1buf, c2w, c2b, c2buf);
  k_conv3<<<dim3((4 * TT) / 256, 128), 256, 0, stream>>>(c2buf, c3w, c3b, h);

  // --- transformer layers ---
  for (int l = 0; l < NLAYERS; ++l) {
    _Float16* wl  = wT + (size_t)l * LSTRIDE;
    const float* bq_l = bq + l * 128;
    const float* bk_l = bk + l * 128;
    const float* bv_l = bv + l * 128;
    const float* bo_l = bo + l * 128;
    // blocks = (M/16) * (N/64) / 8 waves
    const int gem128 = (MROWS / 16) * (128 / 64) / 8;   // 256 blocks
    const int gem512 = (MROWS / 16) * (512 / 64) / 8;   // 1024 blocks

    k_ln<<<MROWS / 8, 256, 0, stream>>>(h, ln1_s + l * 128, ln1_b + l * 128, y16);
    k_gemm<0, 128, 128><<<gem128, 256, 0, stream>>>(y16, wl,          bq_l, qb, nullptr, nullptr, nullptr);
    k_gemm<0, 128, 128><<<gem128, 256, 0, stream>>>(y16, wl + 16384,  bk_l, kb, nullptr, nullptr, nullptr);
    k_gemm<0, 128, 128><<<gem128, 256, 0, stream>>>(y16, wl + 32768,  bv_l, vb, nullptr, nullptr, nullptr);
    k_attn<<<(BB * TT * NHEAD) / 256, 256, 0, stream>>>(qb, kb, vb, Er + (size_t)l * NHEAD * ALEN * HD, a16, 1 << l);
    k_gemm<3, 128, 128><<<gem128, 256, 0, stream>>>(a16, wl + 49152, bo_l, nullptr, nullptr, h, ssum);
    k_ln<<<MROWS / 8, 256, 0, stream>>>(h, ln2_s + l * 128, ln2_b + l * 128, y16);
    k_gemm<1, 512, 128><<<gem512, 256, 0, stream>>>(y16, wl + 65536, b1 + l * 512, nullptr, a1b, nullptr, nullptr);
    k_gemm<2, 128, 512><<<gem128, 256, 0, stream>>>(a1b, wl + 131072, b2 + l * 128, nullptr, nullptr, h, nullptr);
  }

  // --- output heads ---
  k_out<<<MROWS / 256, 256, 0, stream>>>(h, Wout, bout, out);
  k_tpool<<<BB, 128, 0, stream>>>(ssum, mb);
  k_tout<<<(BB * 300 + 255) / 256, 256, 0, stream>>>(mb, Wt, bt, out + (size_t)MROWS * 2);
}
